// MoleculeGenerator_75222057222369
// MI455X (gfx1250) — matrix operand order repaired and checked by reading, not compiled, not run on hardware
//
#include <hip/hip_runtime.h>
#include <hip/hip_bf16.h>
#include <math.h>

// ---------------------------------------------------------------------------
// Problem constants (match reference)
// ---------------------------------------------------------------------------
#define BB   4096
#define TT   100
#define VV   512
#define HH   512
#define LAT  128
#define PDIM 1024
#define G4H  2048           // 4*H

typedef __bf16 bf16_t;
typedef bf16_t v16bf __attribute__((ext_vector_type(16)));
typedef bf16_t v8bf  __attribute__((ext_vector_type(8)));
typedef float  v8f   __attribute__((ext_vector_type(8)));
typedef unsigned int v4u __attribute__((ext_vector_type(4)));
typedef int  v8i __attribute__((ext_vector_type(8)));
typedef int  v4i __attribute__((ext_vector_type(4)));

#define TILE_M 128
#define TILE_N 128
#define TILE_K 32

// ---------------------------------------------------------------------------
// TDM: issue one tensor_load_to_lds moving a 128x32 bf16 tile (row-major,
// row stride = stride_elems) from global memory into LDS at lds_addr.
// D# packing per CDNA5 ISA ch.8 (group0: count/lds/global/type, group1:
// data_size + tensor dims 32x128 + tile dims 32x128 + dim0 stride).
// This toolchain exposes the 6-arg builtin (extra int32x8 group before cpol).
// ---------------------------------------------------------------------------
__device__ __forceinline__ void tdm_load_tile_128x32(unsigned lds_addr,
                                                     const bf16_t* gptr,
                                                     int stride_elems) {
    unsigned long long ga = (unsigned long long)(size_t)gptr;
    v4u g0 = { 1u,                                   // count=1, user descriptor
               lds_addr,                             // LDS byte address
               (unsigned)(ga & 0xffffffffull),       // global_addr[31:0]
               (unsigned)((ga >> 32) & 0x01ffffffull) | (2u << 30) }; // [56:32] | type=2
    v8i g1 = { (int)0x00010000,                      // wg_mask=0, data_size=1 (2 bytes)
               (int)(((unsigned)TILE_K & 0xffffu) << 16),   // tensor_dim0[15:0]=32
               (int)(((unsigned)TILE_M & 0xffffu) << 16),   // dim0 hi=0 | tensor_dim1=128
               (int)(((unsigned)TILE_K) << 16),             // dim1 hi=0 | tile_dim0=32
               (int)TILE_M,                                  // tile_dim1=128, tile_dim2=0
               (int)stride_elems,                            // tensor_dim0_stride[31:0]
               0, 0 };                                       // stride hi, dim1_stride=0
    v4i z4 = { 0, 0, 0, 0 };
    v8i z8 = { 0, 0, 0, 0, 0, 0, 0, 0 };
    __builtin_amdgcn_tensor_load_to_lds(g0, g1, z4, z4, z8, 0);
}

__device__ __forceinline__ unsigned lds_addr_of(const void* p) {
    // generic pointer to LDS: low 32 bits are the LDS byte offset
    return (unsigned)(size_t)p;
}

// ---------------------------------------------------------------------------
// WMMA GEMM: C[M,N] (f32) = A[M,K] (bf16, row-major, lda) * W[N,K]^T (bf16, ldw)
// 256 threads = 8 waves; block tile 128x128; wave tile 32x64 (2x4 frags).
// Tile staging via TDM (double buffered); compute via v_wmma_f32_16x16x32_bf16.
// ---------------------------------------------------------------------------
__global__ __launch_bounds__(256)
void gemm_bf16_wmma(const bf16_t* __restrict__ A, int lda,
                    const bf16_t* __restrict__ W, int ldw,
                    float* __restrict__ C, int ldc,
                    int M, int N, int K)
{
    __shared__ __align__(128) bf16_t As[2][TILE_M][TILE_K];   // 2 x 8 KB
    __shared__ __align__(128) bf16_t Ws[2][TILE_N][TILE_K];   // 2 x 8 KB

    const int tid  = threadIdx.x;
    const int lane = tid & 31;
    const int wave = tid >> 5;              // 0..7
    const int wm   = (wave >> 1) * 32;      // 0,32,64,96
    const int wn   = (wave & 1) * 64;       // 0,64

    const int m0 = blockIdx.y * TILE_M;
    const int n0 = blockIdx.x * TILE_N;

    v8f acc[2][4] = {};                     // 64 f32 accumulator VGPRs

    const int half = (lane < 16) ? 0 : 1;   // K-half select per ISA frag layout
    const int lr   = lane & 15;

    const int nk = K / TILE_K;

    // prologue: DMA first K-tiles of A and W into buffer 0
    if (wave == 0) {
        tdm_load_tile_128x32(lds_addr_of(&As[0][0][0]),
                             A + (size_t)m0 * lda, lda);
        tdm_load_tile_128x32(lds_addr_of(&Ws[0][0][0]),
                             W + (size_t)n0 * ldw, ldw);
    }

    for (int ks = 0; ks < nk; ++ks) {
        const int buf = ks & 1;
        // all waves done reading buf^1 from the previous iteration
        __syncthreads();
        if (wave == 0) {
            if (ks + 1 < nk) {
                const int k1 = (ks + 1) * TILE_K;
                tdm_load_tile_128x32(lds_addr_of(&As[buf ^ 1][0][0]),
                                     A + (size_t)m0 * lda + k1, lda);
                tdm_load_tile_128x32(lds_addr_of(&Ws[buf ^ 1][0][0]),
                                     W + (size_t)n0 * ldw + k1, ldw);
                // in-order pairs: <=2 outstanding => pair(ks) has landed
                __builtin_amdgcn_s_wait_tensorcnt(2);
            } else {
                __builtin_amdgcn_s_wait_tensorcnt(0);
            }
        }
        // publish buf to all waves
        __syncthreads();

        // A fragments: lane<16 -> row M=lr, K {0..7,16..23}; lane>=16 -> K {8..15,24..31}
        union Frag { v16bf v; v8bf h[2]; };
        Frag af[2], bf[4];
#pragma unroll
        for (int i = 0; i < 2; ++i) {
            const bf16_t* p = &As[buf][wm + i * 16 + lr][half * 8];
            af[i].h[0] = *(const v8bf*)p;
            af[i].h[1] = *(const v8bf*)(p + 16);
        }
        // B fragments: lane<16 -> col N=lr, K {0..7,16..23}; lane>=16 -> K {8..15,24..31}
#pragma unroll
        for (int j = 0; j < 4; ++j) {
            const bf16_t* p = &Ws[buf][wn + j * 16 + lr][half * 8];
            bf[j].h[0] = *(const v8bf*)p;
            bf[j].h[1] = *(const v8bf*)(p + 16);
        }
#pragma unroll
        for (int i = 0; i < 2; ++i)
#pragma unroll
            for (int j = 0; j < 4; ++j)
                acc[i][j] = __builtin_amdgcn_wmma_f32_16x16x32_bf16(
                    false, af[i].v, false, bf[j].v,
                    (short)0, acc[i][j], false, false);
    }

    // C/D layout: VGPR r -> M = r (lanes 0-15) or r+8 (lanes 16-31); N = lane&15
    const int crow = (lane < 16) ? 0 : 8;
#pragma unroll
    for (int i = 0; i < 2; ++i)
#pragma unroll
        for (int j = 0; j < 4; ++j)
#pragma unroll
            for (int r = 0; r < 8; ++r) {
                const int row = m0 + wm + i * 16 + crow + r;
                const int col = n0 + wn + j * 16 + lr;
                C[(size_t)row * ldc + col] = acc[i][j][r];
            }
}

// ---------------------------------------------------------------------------
// Elementwise helpers
// ---------------------------------------------------------------------------
__global__ void cvt_f32_bf16(const float* __restrict__ src, bf16_t* __restrict__ dst, int n) {
    int i = blockIdx.x * blockDim.x + threadIdx.x;
    if (i < n) dst[i] = (bf16_t)src[i];
}

// pack [wih1 | whh1] -> Wcat[2048][1024] bf16
__global__ void pack_wcat(const float* __restrict__ wih1, const float* __restrict__ whh1,
                          bf16_t* __restrict__ dst) {
    int i = blockIdx.x * blockDim.x + threadIdx.x;
    if (i >= G4H * 1024) return;
    int n = i >> 10, k = i & 1023;
    float v = (k < HH) ? wih1[n * HH + k] : whh1[n * HH + (k - HH)];
    dst[i] = (bf16_t)v;
}

// wih0T[v][r] = wih0[r][v]   (f32, [VV][G4H])
__global__ void transpose_wih0(const float* __restrict__ wih0, float* __restrict__ wih0T) {
    int i = blockIdx.x * blockDim.x + threadIdx.x;
    if (i >= VV * G4H) return;
    int v = i >> 11, r = i & (G4H - 1);
    wih0T[i] = wih0[r * VV + v];
}

__global__ void add_bias_pair(const float* __restrict__ a, const float* __restrict__ b,
                              float* __restrict__ o, int n) {
    int i = blockIdx.x * blockDim.x + threadIdx.x;
    if (i < n) o[i] = a[i] + b[i];
}

// dst[m*ldd + doff + n] = bf16( act( C[m*ldc + n] + bias[n] ) )
__global__ void bias_act_cvt(const float* __restrict__ C, int ldc,
                             const float* __restrict__ bias,
                             bf16_t* __restrict__ dst, int ldd, int doff,
                             int M, int N, int relu) {
    int i = blockIdx.x * blockDim.x + threadIdx.x;
    if (i >= M * N) return;
    int m = i / N, n = i - m * N;
    float v = C[(size_t)m * ldc + n];
    if (bias) v += bias[n];
    if (relu) v = fmaxf(v, 0.f);
    dst[(size_t)m * ldd + doff + n] = (bf16_t)v;
}

__device__ __forceinline__ float sigf(float x) { return 1.f / (1.f + __expf(-x)); }

// LSTM cell epilogue: gates G[B][2048] (+ optional one-hot gather) -> hcat half
__global__ void lstm_cell_epi(const float* __restrict__ G,
                              const float* __restrict__ bsum,
                              const float* __restrict__ wih0T,   // nullable
                              const int*   __restrict__ tok,     // nullable
                              bf16_t* __restrict__ hcat, int hoff) {
    int i = blockIdx.x * blockDim.x + threadIdx.x;
    if (i >= BB * HH) return;
    int b = i >> 9, j = i & (HH - 1);
    const float* g = G + (size_t)b * G4H;
    float gi = g[j]          + bsum[j];
    float gg = g[j + 2 * HH] + bsum[j + 2 * HH];
    float go = g[j + 3 * HH] + bsum[j + 3 * HH];
    if (wih0T) {
        const float* w = wih0T + (size_t)tok[b] * G4H;
        gi += w[j]; gg += w[j + 2 * HH]; go += w[j + 3 * HH];
    }
    float c = sigf(gi) * tanhf(gg);
    float h = sigf(go) * tanhf(c);
    hcat[(size_t)b * (2 * HH) + hoff + j] = (bf16_t)h;
}

__global__ void init_tok(int* __restrict__ tok) {
    int i = blockIdx.x * blockDim.x + threadIdx.x;
    if (i < BB) tok[i] = VV - 1;               // one-hot at V-1
}

__device__ __forceinline__ float gumbelf(unsigned s, unsigned b, unsigned v) {
    unsigned x = s * 2654435761u ^ b * 2246822519u ^ v * 3266489917u ^ 0x9E3779B9u;
    x ^= x >> 16; x *= 2246822519u; x ^= x >> 13; x *= 3266489917u; x ^= x >> 16;
    float u = (float)(x >> 8) * (1.0f / 16777216.0f) + 1e-7f;
    return -logf(-logf(u));
}

// Gumbel-max categorical sampling: one block per batch row
__global__ __launch_bounds__(256)
void sample_tok(const float* __restrict__ logits, const float* __restrict__ outb,
                int* __restrict__ tok, int* __restrict__ out, int step) {
    __shared__ float sv[256];
    __shared__ int   si[256];
    const int b = blockIdx.x, t = threadIdx.x;
    float best = -INFINITY; int bi = 0;
    for (int v = t; v < VV; v += 256) {
        float key = logits[(size_t)b * VV + v] + outb[v] + gumbelf(step, b, v);
        if (key > best) { best = key; bi = v; }
    }
    sv[t] = best; si[t] = bi;
    __syncthreads();
    for (int s = 128; s > 0; s >>= 1) {
        if (t < s && sv[t + s] > sv[t]) { sv[t] = sv[t + s]; si[t] = si[t + s]; }
        __syncthreads();
    }
    if (t == 0) { tok[b] = si[0]; out[(size_t)b * TT + step] = si[0]; }
}

// ---------------------------------------------------------------------------
// Host-side orchestration
// ---------------------------------------------------------------------------
extern "C" void kernel_launch(void* const* d_in, const int* in_sizes, int n_in,
                              void* d_out, int out_size, void* d_ws, size_t ws_size,
                              hipStream_t stream) {
    const float* pf    = (const float*)d_in[0];
    const float* z     = (const float*)d_in[1];
    const float* pe_w1 = (const float*)d_in[2];
    const float* pe_b1 = (const float*)d_in[3];
    const float* pe_w2 = (const float*)d_in[4];
    const float* pe_b2 = (const float*)d_in[5];
    const float* fc_w  = (const float*)d_in[6];
    const float* fc_b  = (const float*)d_in[7];
    const float* wih0  = (const float*)d_in[8];
    const float* whh0  = (const float*)d_in[9];
    const float* bih0  = (const float*)d_in[10];
    const float* bhh0  = (const float*)d_in[11];
    const float* wih1  = (const float*)d_in[12];
    const float* whh1  = (const float*)d_in[13];
    const float* bih1  = (const float*)d_in[14];
    const float* bhh1  = (const float*)d_in[15];
    const float* out_w = (const float*)d_in[16];
    const float* out_b = (const float*)d_in[17];
    int* out = (int*)d_out;

    // workspace carve (256-byte aligned)
    char* ws = (char*)d_ws;
    size_t off = 0;
    auto alloc = [&](size_t bytes) {
        void* p = ws + off;
        off += (bytes + 255) & ~(size_t)255;
        return p;
    };
    float*  G       = (float*)alloc((size_t)BB * G4H * 4);     // GEMM C / gates / logits
    bf16_t* hcat    = (bf16_t*)alloc((size_t)BB * 2 * HH * 2); // [ha | hb]
    bf16_t* pf_bf   = (bf16_t*)alloc((size_t)BB * PDIM * 2);
    bf16_t* pe1_bf  = (bf16_t*)alloc((size_t)BB * 512 * 2);
    bf16_t* zcat    = (bf16_t*)alloc((size_t)BB * (LAT + 256) * 2);
    bf16_t* w1_bf   = (bf16_t*)alloc((size_t)512 * PDIM * 2);
    bf16_t* w2_bf   = (bf16_t*)alloc((size_t)256 * 512 * 2);
    bf16_t* fcw_bf  = (bf16_t*)alloc((size_t)HH * (LAT + 256) * 2);
    bf16_t* whh0_bf = (bf16_t*)alloc((size_t)G4H * HH * 2);
    bf16_t* wcat_bf = (bf16_t*)alloc((size_t)G4H * 1024 * 2);
    bf16_t* outw_bf = (bf16_t*)alloc((size_t)VV * HH * 2);
    float*  wih0T   = (float*)alloc((size_t)VV * G4H * 4);
    float*  b0      = (float*)alloc((size_t)G4H * 4);
    float*  b1      = (float*)alloc((size_t)G4H * 4);
    int*    tok     = (int*)alloc((size_t)BB * 4);
    (void)ws_size; (void)in_sizes; (void)n_in; (void)out_size;

    auto nb = [](int n) { return (n + 255) / 256; };
    auto cvt = [&](const float* s, bf16_t* d, int n) {
        cvt_f32_bf16<<<nb(n), 256, 0, stream>>>(s, d, n);
    };
    auto gemm = [&](const bf16_t* A, int lda, const bf16_t* W, int ldw,
                    float* C, int ldc, int M, int N, int K) {
        dim3 g(N / TILE_N, M / TILE_M);
        gemm_bf16_wmma<<<g, 256, 0, stream>>>(A, lda, W, ldw, C, ldc, M, N, K);
    };

    // ---- one-time packing (re-done every call; deterministic) ----
    cvt(pf,    pf_bf,   BB * PDIM);
    cvt(pe_w1, w1_bf,   512 * PDIM);
    cvt(pe_w2, w2_bf,   256 * 512);
    cvt(fc_w,  fcw_bf,  HH * (LAT + 256));
    cvt(whh0,  whh0_bf, G4H * HH);
    cvt(out_w, outw_bf, VV * HH);
    pack_wcat<<<nb(G4H * 1024), 256, 0, stream>>>(wih1, whh1, wcat_bf);
    transpose_wih0<<<nb(VV * G4H), 256, 0, stream>>>(wih0, wih0T);
    add_bias_pair<<<nb(G4H), 256, 0, stream>>>(bih0, bhh0, b0, G4H);
    add_bias_pair<<<nb(G4H), 256, 0, stream>>>(bih1, bhh1, b1, G4H);
    init_tok<<<nb(BB), 256, 0, stream>>>(tok);

    // z -> zcat[:, 0:128]
    bias_act_cvt<<<nb(BB * LAT), 256, 0, stream>>>(z, LAT, nullptr, zcat, LAT + 256, 0,
                                                   BB, LAT, 0);
    // pe1 = relu(pf @ pe_w1.T + b1e)
    gemm(pf_bf, PDIM, w1_bf, PDIM, G, 512, BB, 512, PDIM);
    bias_act_cvt<<<nb(BB * 512), 256, 0, stream>>>(G, 512, pe_b1, pe1_bf, 512, 0,
                                                   BB, 512, 1);
    // pe2 = relu(pe1 @ pe_w2.T + b2e) -> zcat[:, 128:384]
    gemm(pe1_bf, 512, w2_bf, 512, G, 256, BB, 256, 512);
    bias_act_cvt<<<nb(BB * 256), 256, 0, stream>>>(G, 256, pe_b2, zcat, LAT + 256, LAT,
                                                   BB, 256, 1);
    // h0 = zcat @ fc_w.T + fc_b  -> both halves of hcat
    gemm(zcat, LAT + 256, fcw_bf, LAT + 256, G, HH, BB, HH, LAT + 256);
    bias_act_cvt<<<nb(BB * HH), 256, 0, stream>>>(G, HH, fc_b, hcat, 2 * HH, 0,
                                                  BB, HH, 0);
    bias_act_cvt<<<nb(BB * HH), 256, 0, stream>>>(G, HH, fc_b, hcat, 2 * HH, HH,
                                                  BB, HH, 0);

    // ---- T sequential decode steps ----
    for (int t = 0; t < TT; ++t) {
        // cell 0: gates = ha @ whh0.T (+ gather of wih0T[tok] + b0 in epilogue)
        gemm(hcat, 2 * HH, whh0_bf, HH, G, G4H, BB, G4H, HH);
        lstm_cell_epi<<<nb(BB * HH), 256, 0, stream>>>(G, b0, wih0T, tok, hcat, 0);
        // cell 1: gates = [ha|hb] @ [wih1|whh1].T + b1
        gemm(hcat, 2 * HH, wcat_bf, 2 * HH, G, G4H, BB, G4H, 2 * HH);
        lstm_cell_epi<<<nb(BB * HH), 256, 0, stream>>>(G, b1, nullptr, nullptr, hcat, HH);
        // logits = hb @ out_w.T  (out_b added in sampler)
        gemm(hcat + HH, 2 * HH, outw_bf, HH, G, VV, BB, VV, HH);
        sample_tok<<<BB, 256, 0, stream>>>(G, out_b, tok, out, t);
    }
}
